// RopeAttention_35356170781271
// MI455X (gfx1250) — compile-verified, matrix operand order checked by reading, not run on hardware
//
#include <hip/hip_runtime.h>
#include <hip/hip_bf16.h>

#define T_SEQ   2048
#define NH      12
#define DH      64
#define DM      768
#define MROWS   4096   /* B*T */
#define WSTRIDE (DM * DM)   /* elements between WQt / WKt / WVt in ws */

typedef __attribute__((ext_vector_type(16))) __bf16 bf16x16;
typedef __attribute__((ext_vector_type(8)))  __bf16 bf16x8;
typedef __attribute__((ext_vector_type(8)))  float  f32x8;

__device__ __forceinline__ unsigned short f2bfu(float f) {
    unsigned u = __builtin_bit_cast(unsigned, f);
    u += 0x7FFFu + ((u >> 16) & 1u);           // round-to-nearest-even
    return (unsigned short)(u >> 16);
}

__device__ __forceinline__ f32x8 wmma_bf16(bf16x16 a, bf16x16 b, f32x8 c) {
    return __builtin_amdgcn_wmma_f32_16x16x32_bf16(
        false, a, false, b, (short)0, c, false, false);
}

// 16x32 bf16 operand tile from uniform base + BYTE offset (folds into saddr+imm form).
__device__ __forceinline__ bf16x16 ld16b(const char* __restrict__ base, unsigned boff) {
    bf16x8 lo = *reinterpret_cast<const bf16x8*>(base + boff);
    bf16x8 hi = *reinterpret_cast<const bf16x8*>(base + boff + 32);
    bf16x16 out;
#pragma unroll
    for (int i = 0; i < 8; ++i) { out[i] = lo[i]; out[i + 8] = hi[i]; }
    return out;
}

// ---------------- conversion kernels ----------------
__global__ void conv_x_kernel(const float* __restrict__ x, unsigned short* __restrict__ y, int n) {
    int i = blockIdx.x * blockDim.x + threadIdx.x;
    if (i < n) y[i] = f2bfu(x[i]);
}

// W (H, DM, DH) -> Wt[n=h*64+d][c], bf16
__global__ void conv_wqkv_kernel(const float* __restrict__ w, unsigned short* __restrict__ y) {
    int i = blockIdx.x * blockDim.x + threadIdx.x;
    if (i >= DM * DM) return;
    int n = i / DM, c = i % DM;
    y[i] = f2bfu(w[(size_t)(n >> 6) * (DM * DH) + (size_t)c * DH + (n & 63)]);
}

// W_O (H, DH, DM) -> WOt[m][k=h*64+d], bf16
__global__ void conv_wo_kernel(const float* __restrict__ w, unsigned short* __restrict__ y) {
    int i = blockIdx.x * blockDim.x + threadIdx.x;
    if (i >= DM * DM) return;
    int m = i / DM, k = i % DM;
    y[i] = f2bfu(w[(size_t)(k >> 6) * (DH * DM) + (size_t)(k & 63) * DM + m]);
}

// ---------------- QKV projection + RoPE ----------------
// grid (MROWS/128, NH), block 256. Wave: 16 rows x 64 cols of head h for Q, K and V.
// Wall = WQt base; WKt at +WSTRIDE, WVt at +2*WSTRIDE (byte immediates fit imm24).
__global__ void qkv_rope_kernel(const unsigned short* __restrict__ Xbf,
                                const unsigned short* __restrict__ Wall,
                                const float* __restrict__ bQ, const float* __restrict__ bK,
                                const float* __restrict__ bV,
                                unsigned short* __restrict__ Qbf,
                                unsigned short* __restrict__ Kbf,
                                unsigned short* __restrict__ Vt) {
    int wave = threadIdx.x >> 5, lane = threadIdx.x & 31;
    int h = blockIdx.y;
    int row0 = blockIdx.x * 128 + wave * 16;
    int l16 = lane & 15, half = lane >> 4;

    const char* Xb = (const char*)Xbf;
    const char* Wb = (const char*)Wall;

    f32x8 qa[4], ka[4], va[4];
#pragma unroll
    for (int nt = 0; nt < 4; ++nt) { qa[nt] = {}; ka[nt] = {}; va[nt] = {}; }

    unsigned ao = ((unsigned)(row0 + l16) * DM + half * 8) * 2;   // byte offsets
    unsigned wo = ((unsigned)(h * 64 + l16) * DM + half * 8) * 2;

    bf16x16 a = ld16b(Xb, ao);
    for (int kt = 0; kt < DM / 32; ++kt) {
        bf16x16 anext;
        if (kt + 1 < DM / 32) anext = ld16b(Xb, ao + 64);
        // stage nt=0 tiles, then pipeline: load nt+1 while multiplying nt
        bf16x16 bq = ld16b(Wb, wo);
        bf16x16 bk = ld16b(Wb, wo + WSTRIDE * 2);
        bf16x16 bv = ld16b(Wb, wo + 2 * WSTRIDE * 2);
#pragma unroll
        for (int nt = 0; nt < 4; ++nt) {
            bf16x16 nbq, nbk, nbv;
            if (nt < 3) {
                nbq = ld16b(Wb, wo + (nt + 1) * 16 * DM * 2);
                nbk = ld16b(Wb, wo + (nt + 1) * 16 * DM * 2 + WSTRIDE * 2);
                nbv = ld16b(Wb, wo + (nt + 1) * 16 * DM * 2 + 2 * WSTRIDE * 2);
            }
            qa[nt] = wmma_bf16(a, bq, qa[nt]);
            ka[nt] = wmma_bf16(a, bk, ka[nt]);
            va[nt] = wmma_bf16(a, bv, va[nt]);
            bq = nbq; bk = nbk; bv = nbv;
        }
        a = anext;
        ao += 64; wo += 64;
    }

#pragma unroll
    for (int nt = 0; nt < 4; ++nt) {
        int d = nt * 16 + l16;
        float bq = bQ[h * 64 + d], bk = bK[h * 64 + d], bv = bV[h * 64 + d];
        int pk = d >> 1;
        // inv_freq = 10000^(-2*pk/64)
        float inv = __expf(-(float)(2 * pk) * (9.210340371976184f / 64.0f));
#pragma unroll
        for (int r = 0; r < 8; ++r) {
            int row = row0 + r + 8 * half;
            int pos = row & (T_SEQ - 1);
            int b = row >> 11;
            float ang = (float)pos * inv;
            float cs = __cosf(ang), sn = __sinf(ang);
            float q = qa[nt][r] + bq;
            float k = ka[nt][r] + bk;
            float qp = __shfl_xor(q, 1, 32);
            float kp = __shfl_xor(k, 1, 32);
            float qr, kr;
            if ((d & 1) == 0) { qr = q * cs - qp * sn; kr = k * cs - kp * sn; }
            else              { qr = qp * sn + q * cs; kr = kp * sn + k * cs; }
            float v = va[nt][r] + bv;
            size_t qoff = (((size_t)b * NH + h) * T_SEQ + pos) * DH + d;
            Qbf[qoff] = f2bfu(qr);
            Kbf[qoff] = f2bfu(kr);
            size_t voff = (((size_t)b * NH + h) * DH + d) * T_SEQ + pos;
            Vt[voff] = f2bfu(v);
        }
    }
}

// ---------------- causal flash attention ----------------
// grid (B*NH, T/128), block 256. Wave: 16 q rows; online softmax over 32-wide k blocks.
__global__ void flash_kernel(const unsigned short* __restrict__ Qbf,
                             const unsigned short* __restrict__ Kbf,
                             const unsigned short* __restrict__ Vt,
                             unsigned short* __restrict__ Obf) {
    __shared__ unsigned short pbuf[8 * 16 * 32];   // per-wave 16x32 bf16 P staging
    int wave = threadIdx.x >> 5, lane = threadIdx.x & 31;
    int l16 = lane & 15, half = lane >> 4;
    int bh = blockIdx.x, b = bh / NH, h = bh % NH;
    int qt = blockIdx.y * 128 + wave * 16;

    const char* Qb = (const char*)(Qbf + ((size_t)bh) * T_SEQ * DH);
    const char* Kb = (const char*)(Kbf + ((size_t)bh) * T_SEQ * DH);
    const char* Vb = (const char*)(Vt  + ((size_t)bh) * DH * T_SEQ);
    unsigned short* psm = pbuf + wave * (16 * 32);

    unsigned qo = ((unsigned)(qt + l16) * DH + half * 8) * 2;
    bf16x16 aq0 = ld16b(Qb, qo);
    bf16x16 aq1 = ld16b(Qb, qo + 64);

    unsigned ko = ((unsigned)l16 * DH + half * 8) * 2;    // += 32*DH*2 per k block
    unsigned vo = ((unsigned)l16 * T_SEQ + half * 8) * 2; // += 32*2 per k block

    f32x8 o[4]; float mrow[8], lrow[8];
#pragma unroll
    for (int dn = 0; dn < 4; ++dn) o[dn] = {};
#pragma unroll
    for (int r = 0; r < 8; ++r) { mrow[r] = -1e30f; lrow[r] = 0.f; }

    int nkb = (qt >> 5) + 1;   // causal: k cols <= qt+15
    for (int kb = 0; kb < nkb; ++kb) {
        int kcol0 = kb * 32;
        if (kb + 1 < nkb) {                     // prefetch next K block
            __builtin_prefetch(Kb + ko + 32 * DH * 2, 0, 0);
            __builtin_prefetch(Kb + ko + 32 * DH * 2 + 16 * DH * 2, 0, 0);
        }
        f32x8 s0 = {}, s1 = {};
        {
            bf16x16 b00 = ld16b(Kb, ko);
            bf16x16 b01 = ld16b(Kb, ko + 64);
            bf16x16 b10 = ld16b(Kb, ko + 16 * DH * 2);
            bf16x16 b11 = ld16b(Kb, ko + 16 * DH * 2 + 64);
            s0 = wmma_bf16(aq0, b00, s0);
            s0 = wmma_bf16(aq1, b01, s0);
            s1 = wmma_bf16(aq0, b10, s1);
            s1 = wmma_bf16(aq1, b11, s1);
        }
        // V tiles issued early: their latency is hidden by the softmax below
        bf16x16 vt0 = ld16b(Vb, vo);
        bf16x16 vt1 = ld16b(Vb, vo + 16 * T_SEQ * 2);
        bf16x16 vt2 = ld16b(Vb, vo + 2 * 16 * T_SEQ * 2);
        bf16x16 vt3 = ld16b(Vb, vo + 3 * 16 * T_SEQ * 2);

        float mb[8];
#pragma unroll
        for (int r = 0; r < 8; ++r) {
            int rowq = qt + r + 8 * half;
            float a0 = s0[r] * 0.125f; if (kcol0 + l16 > rowq)      a0 = -1e30f;
            float a1 = s1[r] * 0.125f; if (kcol0 + 16 + l16 > rowq) a1 = -1e30f;
            s0[r] = a0; s1[r] = a1;
            mb[r] = fmaxf(a0, a1);
        }
#pragma unroll
        for (int msk = 1; msk < 16; msk <<= 1)
#pragma unroll
            for (int r = 0; r < 8; ++r)
                mb[r] = fmaxf(mb[r], __shfl_xor(mb[r], msk, 32));

        float rs[8];
#pragma unroll
        for (int r = 0; r < 8; ++r) {
            float mn = fmaxf(mrow[r], mb[r]);
            float alpha = __expf(mrow[r] - mn);
            mrow[r] = mn;
            float p0 = __expf(s0[r] - mn);
            float p1 = __expf(s1[r] - mn);
            int mr = r + 8 * half;
            psm[mr * 32 + l16]      = f2bfu(p0);
            psm[mr * 32 + 16 + l16] = f2bfu(p1);
            rs[r] = p0 + p1;
#pragma unroll
            for (int dn = 0; dn < 4; ++dn) o[dn][r] *= alpha;
            lrow[r] = lrow[r] * alpha;
        }
#pragma unroll
        for (int msk = 1; msk < 16; msk <<= 1)
#pragma unroll
            for (int r = 0; r < 8; ++r)
                rs[r] += __shfl_xor(rs[r], msk, 32);
#pragma unroll
        for (int r = 0; r < 8; ++r) lrow[r] += rs[r];

        asm volatile("s_wait_dscnt 0" ::: "memory");
        // re-read P in A layout (wave-private LDS; DS ops in order within a wave)
        bf16x16 pa;
        {
            const unsigned short* rowp = psm + l16 * 32 + half * 8;
            bf16x8 lo = *reinterpret_cast<const bf16x8*>(rowp);
            bf16x8 hi = *reinterpret_cast<const bf16x8*>(rowp + 16);
#pragma unroll
            for (int i = 0; i < 8; ++i) { pa[i] = lo[i]; pa[i + 8] = hi[i]; }
        }
        o[0] = wmma_bf16(pa, vt0, o[0]);
        o[1] = wmma_bf16(pa, vt1, o[1]);
        o[2] = wmma_bf16(pa, vt2, o[2]);
        o[3] = wmma_bf16(pa, vt3, o[3]);

        ko += 32 * DH * 2;
        vo += 32 * 2;
    }

#pragma unroll
    for (int r = 0; r < 8; ++r) {
        float invl = 1.0f / lrow[r];
        int rowg = b * T_SEQ + qt + r + 8 * half;
#pragma unroll
        for (int dn = 0; dn < 4; ++dn) {
            int d = dn * 16 + l16;
            Obf[(size_t)rowg * DM + h * 64 + d] = f2bfu(o[dn][r] * invl);
        }
    }
}

// ---------------- output projection ----------------
// grid (MROWS/128, DM/64), block 256.
__global__ void outproj_kernel(const unsigned short* __restrict__ Obf,
                               const unsigned short* __restrict__ WOt,
                               const float* __restrict__ bO,
                               float* __restrict__ out) {
    int wave = threadIdx.x >> 5, lane = threadIdx.x & 31;
    int l16 = lane & 15, half = lane >> 4;
    int row0 = blockIdx.x * 128 + wave * 16;
    int n0blk = blockIdx.y * 64;

    const char* Ob = (const char*)Obf;
    const char* Wb = (const char*)WOt;

    f32x8 acc[4];
#pragma unroll
    for (int nt = 0; nt < 4; ++nt) acc[nt] = {};

    unsigned ao = ((unsigned)(row0 + l16) * DM + half * 8) * 2;
    unsigned wo = ((unsigned)(n0blk + l16) * DM + half * 8) * 2;

    bf16x16 a = ld16b(Ob, ao);
    for (int kt = 0; kt < DM / 32; ++kt) {
        bf16x16 anext;
        if (kt + 1 < DM / 32) anext = ld16b(Ob, ao + 64);
        bf16x16 w = ld16b(Wb, wo);
#pragma unroll
        for (int nt = 0; nt < 4; ++nt) {
            bf16x16 nw;
            if (nt < 3) nw = ld16b(Wb, wo + (nt + 1) * 16 * DM * 2);
            acc[nt] = wmma_bf16(a, w, acc[nt]);
            w = nw;
        }
        a = anext;
        ao += 64; wo += 64;
    }
#pragma unroll
    for (int nt = 0; nt < 4; ++nt) {
        int n = n0blk + nt * 16 + l16;
        float bias = bO[n];
#pragma unroll
        for (int r = 0; r < 8; ++r)
            out[(size_t)(row0 + r + 8 * half) * DM + n] = acc[nt][r] + bias;
    }
}

extern "C" void kernel_launch(void* const* d_in, const int* in_sizes, int n_in,
                              void* d_out, int out_size, void* d_ws, size_t ws_size,
                              hipStream_t stream) {
    const float* x  = (const float*)d_in[0];
    const float* WQ = (const float*)d_in[1];
    const float* bQ = (const float*)d_in[2];
    const float* WK = (const float*)d_in[3];
    const float* bK = (const float*)d_in[4];
    const float* WV = (const float*)d_in[5];
    const float* bV = (const float*)d_in[6];
    const float* WO = (const float*)d_in[7];
    const float* bO = (const float*)d_in[8];
    float* out = (float*)d_out;

    char* ws = (char*)d_ws;
    const size_t SZ_X   = (size_t)MROWS * DM * 2;
    const size_t SZ_W   = (size_t)DM * DM * 2;
    const size_t SZ_QKV = (size_t)2 * NH * T_SEQ * DH * 2;
    unsigned short* Xbf = (unsigned short*)(ws);
    unsigned short* WQt = (unsigned short*)(ws + SZ_X);            // WKt, WVt contiguous after
    unsigned short* WKt = (unsigned short*)(ws + SZ_X + SZ_W);
    unsigned short* WVt = (unsigned short*)(ws + SZ_X + 2 * SZ_W);
    unsigned short* WOt = (unsigned short*)(ws + SZ_X + 3 * SZ_W);
    unsigned short* Qbf = (unsigned short*)(ws + SZ_X + 4 * SZ_W);
    unsigned short* Kbf = (unsigned short*)(ws + SZ_X + 4 * SZ_W + SZ_QKV);
    unsigned short* Vt  = (unsigned short*)(ws + SZ_X + 4 * SZ_W + 2 * SZ_QKV);
    unsigned short* Obf = (unsigned short*)(ws + SZ_X + 4 * SZ_W + 3 * SZ_QKV);

    const int nx = MROWS * DM;
    conv_x_kernel<<<(nx + 255) / 256, 256, 0, stream>>>(x, Xbf, nx);
    const int nw = DM * DM;
    conv_wqkv_kernel<<<(nw + 255) / 256, 256, 0, stream>>>(WQ, WQt);
    conv_wqkv_kernel<<<(nw + 255) / 256, 256, 0, stream>>>(WK, WKt);
    conv_wqkv_kernel<<<(nw + 255) / 256, 256, 0, stream>>>(WV, WVt);
    conv_wo_kernel<<<(nw + 255) / 256, 256, 0, stream>>>(WO, WOt);

    qkv_rope_kernel<<<dim3(MROWS / 128, NH), 256, 0, stream>>>(
        Xbf, WQt, bQ, bK, bV, Qbf, Kbf, Vt);

    flash_kernel<<<dim3(2 * NH, T_SEQ / 128), 256, 0, stream>>>(Qbf, Kbf, Vt, Obf);

    outproj_kernel<<<dim3(MROWS / 128, DM / 64), 256, 0, stream>>>(Obf, WOt, bO, out);
}
